// Decoder_3058016715187
// MI455X (gfx1250) — matrix operand order repaired and checked by reading, not compiled, not run on hardware
//
#include <hip/hip_runtime.h>

// ---------------------------------------------------------------------------
// GRU decoder for MI455X (gfx1250, wave32, WMMA).
//   B=64, T=64, V=32000, DE=DC=512, DD=1024  ->  K is always 1024.
//   bf16 WMMA GEMMs (gi hoisted out of scan, serial gh per step, big logits
//   GEMM), fp32 accumulate, fused gate VALU kernel.
// ---------------------------------------------------------------------------

typedef __attribute__((ext_vector_type(16))) __bf16 v16bf;
typedef __attribute__((ext_vector_type(8)))  float  v8f;

#define B_   64
#define T_   64
#define V_   32000
#define DE_  512
#define DC_  512
#define DD_  1024
#define K_   1024      // DE+DC == DD == 1024 for every GEMM here
#define N3_  3072      // 3*DD

static __device__ __forceinline__ unsigned short f2bf(float f) {
    // round-to-nearest-even fp32 -> bf16
    unsigned u = __float_as_uint(f);
    u += 0x7FFFu + ((u >> 16) & 1u);
    return (unsigned short)(u >> 16);
}

// ---------------------------------------------------------------------------
// fp32 -> bf16 weight conversion
// ---------------------------------------------------------------------------
__global__ __launch_bounds__(256) void k_cvt_bf16(const float* __restrict__ s,
                                                  unsigned short* __restrict__ d, int n) {
    int i = blockIdx.x * 256 + threadIdx.x;
    if (i < n) d[i] = f2bf(s[i]);
}

// ---------------------------------------------------------------------------
// Build X[t*B+b][1024] = concat(emb[token(t,b)], context[b]) in bf16.
// token(0,b)=bos, token(t,b)=labels[b][t-1].
// ---------------------------------------------------------------------------
__global__ __launch_bounds__(256) void k_build_x(const float* __restrict__ emb,
                                                 const float* __restrict__ ctx,
                                                 const int* __restrict__ labels,
                                                 const int* __restrict__ bos,
                                                 unsigned short* __restrict__ xbf) {
    int idx = blockIdx.x * 256 + threadIdx.x;   // [0, 4096*1024)
    int d   = idx & (K_ - 1);
    int row = idx >> 10;                         // t*B + b
    int t = row >> 6, b = row & 63;
    float v;
    if (d < DE_) {
        int tok = (t == 0) ? bos[0] : labels[b * T_ + (t - 1)];
        v = emb[(size_t)tok * DE_ + d];
    } else {
        v = ctx[b * DC_ + (d - DE_)];
    }
    xbf[idx] = f2bf(v);
}

// ---------------------------------------------------------------------------
// h0 = broadcast(init)
// ---------------------------------------------------------------------------
__global__ __launch_bounds__(256) void k_init_h(const float* __restrict__ init,
                                                float* __restrict__ h,
                                                unsigned short* __restrict__ hbf) {
    int idx = blockIdx.x * 256 + threadIdx.x;   // [0, 64*1024)
    float v = init[idx & (DD_ - 1)];
    h[idx]   = v;
    hbf[idx] = f2bf(v);
}

// ---------------------------------------------------------------------------
// WMMA GEMM: C[M][N] = A[M][K=1024] * W[N][K=1024]^T + bias[N]
//   A, W: bf16 row-major. Workgroup = 256 threads (8 waves).
//   Block tile: (64*WM) x 128.  Wave w: m-group = w>>1 (WM mtiles of 16 rows),
//   n-quad = w&1 (4 x 16 cols).  WM=2 for the big GEMMs (8 WMMA per 12 ds
//   loads), WM=1 for the serial gh step (M=64 exactly).
//   LDS row stride padded to 40 ushorts (80B = 20 banks) so consecutive-row
//   fragment reads spread across bank groups; 16B alignment preserved.
//   MODE 0: out[row*ldo + col] (fp32, row = t*B+b order kept)
//   MODE 1: logits permute: row=t*B+b  ->  out[(b*T+t)*V + col]
// ---------------------------------------------------------------------------
template <int MODE, int WM>
__global__ __launch_bounds__(256) void k_gemm_bf16(const unsigned short* __restrict__ A,
                                                   const unsigned short* __restrict__ W,
                                                   const float* __restrict__ bias,
                                                   float* __restrict__ out, int ldo) {
    constexpr int BM  = 64 * WM;
    constexpr int LDA = 40;                       // padded ushort stride per 32-K row
    __shared__ __align__(16) unsigned short lA[BM * LDA];
    __shared__ __align__(16) unsigned short lB[128 * LDA];

    const int tid   = threadIdx.x;
    const int lane  = tid & 31;
    const int w     = tid >> 5;
    const int mgrp  = w >> 1;                 // 0..3
    const int nq    = w & 1;                  // 0..1
    const int mbase = blockIdx.y * BM;
    const int nbase = blockIdx.x * 128;
    const int l16   = lane & 15;
    const int hiA   = (lane < 16) ? 0 : 8;    // ISA A-frag K split (16-bit A 16x32)
    const int hiB   = (lane < 16) ? 0 : 8;    // B-frag K split (same order as A)

    v8f acc[WM][4] = {};

    // staging: thread -> (row = tid/4 [+64], 16B chunk = tid%4)
    const int rS = tid >> 2;                  // 0..63
    const int cS = (tid & 3) * 8;             // ushort offset of 16B chunk

    for (int kk = 0; kk < K_; kk += 32) {
        // stage A ((64*WM)x32) and B (128x32) tiles into LDS
#pragma unroll
        for (int mi = 0; mi < WM; ++mi) {
            *(uint4*)&lA[(rS + 64 * mi) * LDA + cS] =
                *(const uint4*)(A + (size_t)(mbase + rS + 64 * mi) * K_ + kk + cS);
        }
        *(uint4*)&lB[rS * LDA + cS] =
            *(const uint4*)(W + (size_t)(nbase + rS) * K_ + kk + cS);
        *(uint4*)&lB[(64 + rS) * LDA + cS] =
            *(const uint4*)(W + (size_t)(nbase + 64 + rS) * K_ + kk + cS);
        __syncthreads();

        // A fragments: row = mgrp*WM*16 + mi*16 + l16;
        // elements K = hiA+0..7 and hiA+16..23 (ISA 16-bit A 16x32 layout)
        union FragU { uint4 u[2]; v16bf v; };
        FragU fa[WM];
#pragma unroll
        for (int mi = 0; mi < WM; ++mi) {
            const int ar = (mgrp * WM + mi) * 16 + l16;
            fa[mi].u[0] = *(const uint4*)&lA[ar * LDA + hiA];
            fa[mi].u[1] = *(const uint4*)&lA[ar * LDA + hiA + 16];
        }

#pragma unroll
        for (int j = 0; j < 4; ++j) {
            // B fragment: col = (nq*4+j)*16 + l16; K = hiB+0..7 and hiB+16..23
            FragU fb;
            const int br = (nq * 4 + j) * 16 + l16;
            fb.u[0] = *(const uint4*)&lB[br * LDA + hiB];
            fb.u[1] = *(const uint4*)&lB[br * LDA + hiB + 16];
#pragma unroll
            for (int mi = 0; mi < WM; ++mi) {
                acc[mi][j] = __builtin_amdgcn_wmma_f32_16x16x32_bf16(
                    false, fa[mi].v, false, fb.v, (short)0, acc[mi][j], false, false);
            }
        }
        __syncthreads();
    }

    // Epilogue: C layout -> lane L: col = L%16, rows r + (L<16?0:8), r=0..7
#pragma unroll
    for (int mi = 0; mi < WM; ++mi) {
#pragma unroll
        for (int j = 0; j < 4; ++j) {
            const int col = nbase + (nq * 4 + j) * 16 + l16;
            const float bv = bias[col];
#pragma unroll
            for (int r = 0; r < 8; ++r) {
                const int row = mbase + (mgrp * WM + mi) * 16 + r + ((lane < 16) ? 0 : 8);
                const float v = acc[mi][j][r] + bv;
                if (MODE == 0) {
                    out[(size_t)row * ldo + col] = v;
                } else {
                    const int t = row >> 6, b = row & 63;  // row = t*B+b
                    out[(size_t)(b * T_ + t) * V_ + col] = v;
                }
            }
        }
    }
}

// ---------------------------------------------------------------------------
// Fused GRU gates for step t (PyTorch order r,z,n):
//   r = sig(gi_r+gh_r); z = sig(gi_z+gh_z); n = tanh(gi_n + r*gh_n)
//   h' = (1-z)*n + z*h ;  also emit bf16 h' and states[t] for next GEMMs.
// ---------------------------------------------------------------------------
__global__ __launch_bounds__(256) void k_gru_gate(const float* __restrict__ Gx,
                                                  const float* __restrict__ gh,
                                                  float* __restrict__ h,
                                                  unsigned short* __restrict__ hbf,
                                                  unsigned short* __restrict__ states,
                                                  int t) {
    int idx = blockIdx.x * 256 + threadIdx.x;   // [0, 64*1024)
    int b = idx >> 10, j = idx & (DD_ - 1);
    const float* gi = Gx + (size_t)(t * B_ + b) * N3_;
    const float* gg = gh + (size_t)b * N3_;
    float ir = gi[j], iz = gi[j + DD_], in = gi[j + 2 * DD_];
    float hr = gg[j], hz = gg[j + DD_], hn = gg[j + 2 * DD_];
    float r = 1.0f / (1.0f + __expf(-(ir + hr)));
    float z = 1.0f / (1.0f + __expf(-(iz + hz)));
    float n = tanhf(in + r * hn);
    float ho = h[idx];
    float v = (1.0f - z) * n + z * ho;
    h[idx] = v;
    unsigned short bv = f2bf(v);
    hbf[idx] = bv;
    states[(size_t)(t * B_ + b) * K_ + j] = bv;
}

// ---------------------------------------------------------------------------
// Host launcher
// ---------------------------------------------------------------------------
extern "C" void kernel_launch(void* const* d_in, const int* in_sizes, int n_in,
                              void* d_out, int out_size, void* d_ws, size_t ws_size,
                              hipStream_t stream) {
    const float* ctx    = (const float*)d_in[0];
    const int*   labels = (const int*)d_in[1];
    const float* emb    = (const float*)d_in[2];
    const float* W_ih   = (const float*)d_in[3];
    const float* b_ih   = (const float*)d_in[4];
    const float* W_hh   = (const float*)d_in[5];
    const float* b_hh   = (const float*)d_in[6];
    const float* initv  = (const float*)d_in[7];
    const float* W_out  = (const float*)d_in[8];
    const float* b_out  = (const float*)d_in[9];
    const int*   bos    = (const int*)d_in[10];
    float* logits = (float*)d_out;

    char* ws = (char*)d_ws;
    size_t off = 0;
    auto alloc = [&](size_t bytes) -> char* {
        char* p = ws + off;
        off = (off + bytes + 255) & ~(size_t)255;
        return p;
    };
    unsigned short* Xbf   = (unsigned short*)alloc((size_t)T_ * B_ * K_ * 2);     //  8 MB
    unsigned short* Wihb  = (unsigned short*)alloc((size_t)N3_ * K_ * 2);         //  6 MB
    unsigned short* Whhb  = (unsigned short*)alloc((size_t)N3_ * K_ * 2);         //  6 MB
    unsigned short* Woutb = (unsigned short*)alloc((size_t)V_ * K_ * 2);          // 65 MB
    float*          Gx    = (float*)alloc((size_t)T_ * B_ * N3_ * 4);             // 50 MB
    float*          ghb   = (float*)alloc((size_t)B_ * N3_ * 4);
    float*          h     = (float*)alloc((size_t)B_ * DD_ * 4);
    unsigned short* hbf   = (unsigned short*)alloc((size_t)B_ * DD_ * 2);
    unsigned short* Sbf   = (unsigned short*)alloc((size_t)T_ * B_ * K_ * 2);     //  8 MB

    // 1) weights fp32 -> bf16 (once per call; deterministic)
    k_cvt_bf16<<<(N3_ * K_ + 255) / 256, 256, 0, stream>>>(W_ih, Wihb, N3_ * K_);
    k_cvt_bf16<<<(N3_ * K_ + 255) / 256, 256, 0, stream>>>(W_hh, Whhb, N3_ * K_);
    k_cvt_bf16<<<(V_ * K_ + 255) / 256, 256, 0, stream>>>(W_out, Woutb, V_ * K_);

    // 2) embedding gather + context concat -> bf16 X[t*B+b][1024]
    k_build_x<<<(T_ * B_ * K_) / 256, 256, 0, stream>>>(emb, ctx, labels, bos, Xbf);

    // 3) h0
    k_init_h<<<(B_ * DD_) / 256, 256, 0, stream>>>(initv, h, hbf);

    // 4) gi for ALL timesteps (hoisted out of scan): [4096,1024]x[1024,3072]
    k_gemm_bf16<0, 2><<<dim3(N3_ / 128, (T_ * B_) / 128), 256, 0, stream>>>(
        Xbf, Wihb, b_ih, Gx, N3_);

    // 5) serial recurrence: per step gh GEMM [64,1024]x[1024,3072] + fused gates
    for (int t = 0; t < T_; ++t) {
        k_gemm_bf16<0, 1><<<dim3(N3_ / 128, 1), 256, 0, stream>>>(hbf, Whhb, b_hh, ghb, N3_);
        k_gru_gate<<<(B_ * DD_) / 256, 256, 0, stream>>>(Gx, ghb, h, hbf, Sbf, t);
    }

    // 6) logits: [4096,1024]x[1024,32000], permuted store to [b,t,v] fp32
    k_gemm_bf16<1, 2><<<dim3(V_ / 128, (T_ * B_) / 128), 256, 0, stream>>>(
        Sbf, Woutb, b_out, logits, V_);
}
